// ResLSTM_54778012893249
// MI455X (gfx1250) — compile-verified, matrix operand order checked by reading, not run on hardware
//
#include <hip/hip_runtime.h>
#include <math.h>

// ---------------------------------------------------------------- constants
#define DD     512          // hidden dim
#define TT     2048         // sequence length
#define BB     16           // batch (== WMMA M)
#define KT     (DD/32)      // 16 k-tiles of 32 along D
#define NWG    32           // persistent workgroups per layer kernel
#define SLICE  16           // hidden units per WG (== WMMA N tile)
#define THREADS 256         // 8 waves (wave32)

typedef __attribute__((ext_vector_type(16))) _Float16 v16h;
typedef __attribute__((ext_vector_type(8)))  float    v8f;

// -------------------------------------------------- WMMA packed-layout math
// A operand (16x32 f16): lane holds 16 halves. For k in [0,32):
//   grp = k>>3 ; lane = m + 16*(grp&1) ; slot = (k&7) + 8*(grp>>1)
// B operand (32x16 f16): same mapping with m -> n (column).
// Packed tile = 32 lanes x 16 halves, contiguous per lane (32B/lane).
__device__ __forceinline__ void kslot(int kk, int& laneoff, int& slot) {
    int grp = kk >> 3;
    laneoff = (grp & 1) << 4;
    slot    = (kk & 7) + ((grp >> 1) << 3);
}
// Packed A sequence buffer: [rowtile=t][ktile][lane][slot]
__device__ __forceinline__ size_t packA_off(int rt, int kt, int lane, int slot) {
    return ((((size_t)rt * KT + kt) * 32 + lane) << 4) + slot;
}

// ---------------------------------------------------------------- packers
__global__ void pack_x_kernel(const float* __restrict__ x, _Float16* __restrict__ Xp) {
    size_t tid = (size_t)blockIdx.x * blockDim.x + threadIdx.x;
    if (tid >= (size_t)BB * DD * TT) return;
    int t = (int)(tid % TT);
    int d = (int)((tid / TT) % DD);
    int b = (int)(tid / ((size_t)TT * DD));
    float v = x[((size_t)b * DD + d) * TT + t];
    int kt = d >> 5, kk = d & 31, lo, slot;
    kslot(kk, lo, slot);
    Xp[packA_off(t, kt, b + lo, slot)] = (_Float16)v;
}

__global__ void pack_w_kernel(const float* __restrict__ W, _Float16* __restrict__ Wp) {
    int tid = blockIdx.x * blockDim.x + threadIdx.x;   // 4D*D = 1,048,576 elements
    int k = tid % DD, n = tid / DD;                    // W[n][k], n = gate column
    float v = W[tid];
    int nt = n >> 4, nl = n & 15;
    int kt = k >> 5, kk = k & 31, lo, slot;
    kslot(kk, lo, slot);
    Wp[((((size_t)nt * KT + kt) * 32 + (nl + lo)) << 4) + slot] = (_Float16)v;
}

__global__ void bias_kernel(const float* __restrict__ bi, const float* __restrict__ bh,
                            float* __restrict__ bo) {
    int tid = blockIdx.x * blockDim.x + threadIdx.x;
    if (tid < 4 * DD) bo[tid] = bi[tid] + bh[tid];
}

__global__ void init_ctr_kernel(int* __restrict__ ctr) {
    if (threadIdx.x < 2) ctr[threadIdx.x] = 0;
}

// ------------------------------------------------------- persistent LSTM layer
// WG wg owns hidden units [16*wg, 16*wg+16). Its 4 gate column blocks are
// global ntiles {g*32 + wg}. Waves 0-3: x-projection (W_ih), waves 4-7:
// h-projection (W_hh); wave's gate = wave&3. One 16x16 C tile per wave,
// K=512 -> 16 v_wmma_f32_16x16x32_f16 per projection per step.
__global__ void __launch_bounds__(THREADS)
lstm_layer_kernel(const _Float16* __restrict__ XpIn,   // packed input sequence
                  _Float16* __restrict__ XpOut,        // packed h sequence (written)
                  const _Float16* __restrict__ WihP,
                  const _Float16* __restrict__ WhhP,
                  const float* __restrict__ bias,      // b_ih + b_hh
                  float* __restrict__ yOut,            // f32 h (layer 1 only), or null
                  int* __restrict__ ctr)
{
    __shared__ _Float16 ldsW[2][4][KT][512];   // 128 KB: [ih/hh][gate][ktile][lane*16+slot]
    __shared__ float    slab[2][4][BB][SLICE]; // 8 KB gate pre-activations

    const int wg   = blockIdx.x;
    const int tid  = threadIdx.x;
    const int wave = tid >> 5;
    const int lane = tid & 31;
    const int w_sel = wave >> 2;   // 0 = W_ih (x part), 1 = W_hh (h part)
    const int g_sel = wave & 3;    // gate: i, f, g, o

    // ---- stage this WG's weight tiles into LDS (once, resident all 2048 steps)
    {
        const _Float16* src0[2] = { WihP, WhhP };
        for (int w = 0; w < 2; ++w)
            for (int g = 0; g < 4; ++g) {
                const float4* s = (const float4*)(src0[w] + (size_t)(g * 32 + wg) * KT * 512);
                float4* d = (float4*)&ldsW[w][g][0][0];   // 16 KB = 1024 float4
                for (int i = tid; i < 1024; i += THREADS) d[i] = s[i];
            }
    }
    __syncthreads();

    const int   cN     = lane & 15;                 // column within 16-wide tile
    const int   cMbase = (lane >> 4) << 3;          // M = r + cMbase in C layout
    const float bias_v = (w_sel == 0) ? bias[g_sel * DD + wg * SLICE + cN] : 0.0f;

    // combine-phase ownership: one (batch, hidden) cell per thread, c in a register
    const int cb = tid & 15;        // batch
    const int chd = tid >> 4;       // hidden unit within slice
    float c_reg = 0.0f;

    for (int t = 0; t < TT; ++t) {
        v8f acc = {};
        if (w_sel == 0) {
#pragma unroll
            for (int r = 0; r < 8; ++r) acc[r] = bias_v;
        }
        const bool active = (w_sel == 0) || (t > 0);
        if (active) {
            const _Float16* Abase = (w_sel == 0)
                ? (XpIn  + (size_t)t       * KT * 512)
                : (XpOut + (size_t)(t - 1) * KT * 512);
#pragma unroll
            for (int kt = 0; kt < KT; ++kt) {
                v16h a = *(const v16h*)(Abase + kt * 512 + lane * 16);
                v16h b = *(const v16h*)(&ldsW[w_sel][g_sel][kt][lane * 16]);
                acc = __builtin_amdgcn_wmma_f32_16x16x32_f16(
                        false, a, false, b, (short)0, acc, false, false);
            }
            // warm L2/L0 for the next step's A tile while we sit in the barrier
            __builtin_prefetch((const void*)(Abase + KT * 512 + lane * 16), 0, 1);
        }
        // scatter C tile (lane r -> M = r + cMbase, N = cN) into the gate slab
#pragma unroll
        for (int r = 0; r < 8; ++r) slab[w_sel][g_sel][cMbase + r][cN] = acc[r];
        __syncthreads();

        // ---- elementwise gate combine: c' = sig(f)*c + sig(i)*tanh(g); h = sig(o)*tanh(c')
        {
            float gi = slab[0][0][cb][chd] + slab[1][0][cb][chd];
            float gf = slab[0][1][cb][chd] + slab[1][1][cb][chd];
            float gg = slab[0][2][cb][chd] + slab[1][2][cb][chd];
            float go = slab[0][3][cb][chd] + slab[1][3][cb][chd];
            float i_ = 1.0f / (1.0f + __expf(-gi));
            float f_ = 1.0f / (1.0f + __expf(-gf));
            float o_ = 1.0f / (1.0f + __expf(-go));
            c_reg = f_ * c_reg + i_ * tanhf(gg);
            float h_ = o_ * tanhf(c_reg);

            int kglob = wg * SLICE + chd;
            int kt = kglob >> 5, kk = kglob & 31, lo, slot;
            kslot(kk, lo, slot);
            XpOut[packA_off(t, kt, cb + lo, slot)] = (_Float16)h_;
            if (yOut) yOut[((size_t)t * BB + cb) * DD + kglob] = h_;
        }
        __threadfence();
        __syncthreads();
        // ---- grid-wide step barrier (monotonic counter, L2 resident)
        if (tid == 0) {
            atomicAdd(ctr, 1);
            const int target = NWG * (t + 1);
            while (__hip_atomic_load(ctr, __ATOMIC_ACQUIRE, __HIP_MEMORY_SCOPE_AGENT) < target) {
                __builtin_amdgcn_s_sleep(1);
            }
        }
        __syncthreads();
    }
}

// -------------------------------------------------------------- epilogue
__global__ void epilogue_kernel(const float* __restrict__ y1, const float* __restrict__ x,
                                float* __restrict__ out) {
    size_t tid = (size_t)blockIdx.x * blockDim.x + threadIdx.x;
    if (tid >= (size_t)BB * DD * TT) return;
    int t = (int)(tid % TT);
    int d = (int)((tid / TT) % DD);
    int b = (int)(tid / ((size_t)TT * DD));
    out[tid] = y1[((size_t)t * BB + b) * DD + d] + x[tid];
}

// -------------------------------------------------------------- launcher
extern "C" void kernel_launch(void* const* d_in, const int* in_sizes, int n_in,
                              void* d_out, int out_size, void* d_ws, size_t ws_size,
                              hipStream_t stream) {
    const float* x     = (const float*)d_in[0];
    const float* w_ih0 = (const float*)d_in[1];
    const float* w_hh0 = (const float*)d_in[2];
    const float* b_ih0 = (const float*)d_in[3];
    const float* b_hh0 = (const float*)d_in[4];
    const float* w_ih1 = (const float*)d_in[5];
    const float* w_hh1 = (const float*)d_in[6];
    const float* b_ih1 = (const float*)d_in[7];
    const float* b_hh1 = (const float*)d_in[8];

    char* ws = (char*)d_ws;
    size_t off = 0;
    auto take = [&](size_t bytes) { char* p = ws + off; off = (off + bytes + 255) & ~(size_t)255; return p; };

    const size_t seqPackBytes = (size_t)TT * KT * 512 * sizeof(_Float16);   // 32 MB
    const size_t wPackBytes   = (size_t)128 * KT * 512 * sizeof(_Float16);  // 2 MB

    _Float16* Xp0   = (_Float16*)take(seqPackBytes);  // packed x, later reused as layer-1 h
    _Float16* Xp1   = (_Float16*)take(seqPackBytes);  // layer-0 h == layer-1 input
    _Float16* Wih0p = (_Float16*)take(wPackBytes);
    _Float16* Whh0p = (_Float16*)take(wPackBytes);
    _Float16* Wih1p = (_Float16*)take(wPackBytes);
    _Float16* Whh1p = (_Float16*)take(wPackBytes);
    float*    bias0 = (float*)take(4 * DD * sizeof(float));
    float*    bias1 = (float*)take(4 * DD * sizeof(float));
    float*    y1    = (float*)take((size_t)BB * TT * DD * sizeof(float));   // 64 MB
    int*      ctr   = (int*)take(256);

    const size_t nelem = (size_t)BB * DD * TT;
    const int    packXBlocks = (int)((nelem + 255) / 256);
    const int    packWBlocks = (4 * DD * DD) / 256;

    pack_x_kernel<<<packXBlocks, 256, 0, stream>>>(x, Xp0);
    pack_w_kernel<<<packWBlocks, 256, 0, stream>>>(w_ih0, Wih0p);
    pack_w_kernel<<<packWBlocks, 256, 0, stream>>>(w_hh0, Whh0p);
    pack_w_kernel<<<packWBlocks, 256, 0, stream>>>(w_ih1, Wih1p);
    pack_w_kernel<<<packWBlocks, 256, 0, stream>>>(w_hh1, Whh1p);
    bias_kernel<<<(4 * DD) / 256, 256, 0, stream>>>(b_ih0, b_hh0, bias0);
    bias_kernel<<<(4 * DD) / 256, 256, 0, stream>>>(b_ih1, b_hh1, bias1);
    init_ctr_kernel<<<1, 32, 0, stream>>>(ctr);

    lstm_layer_kernel<<<NWG, THREADS, 0, stream>>>(Xp0, Xp1, Wih0p, Whh0p, bias0, nullptr, ctr + 0);
    lstm_layer_kernel<<<NWG, THREADS, 0, stream>>>(Xp1, Xp0, Wih1p, Whh1p, bias1, y1,     ctr + 1);

    epilogue_kernel<<<packXBlocks, 256, 0, stream>>>(y1, x, (float*)d_out);
}
